// LlamaAttentionFused_30227979829922
// MI455X (gfx1250) — compile-verified, matrix operand order checked by reading, not run on hardware
//
#include <hip/hip_runtime.h>
#include <hip/hip_bf16.h>
#include <cstdint>

typedef __bf16 bf16;
typedef __attribute__((ext_vector_type(16))) __bf16 v16bf;
typedef __attribute__((ext_vector_type(8)))  __bf16 v8bf;
typedef __attribute__((ext_vector_type(8)))  float  v8f;

#define S_LEN 2048
#define HDIM  4096
#define NHEAD 32
#define NKV   8
#define HD    128
#define QSZ   4096
#define KVSZ  1024
#define QKVSZ 6144

__device__ __forceinline__ v16bf ldfrag(const bf16* p) {
  v8bf a = *(const v8bf*)p;
  v8bf b = *(const v8bf*)(p + 16);
  return __builtin_shufflevector(a, b, 0,1,2,3,4,5,6,7,8,9,10,11,12,13,14,15);
}

__device__ __forceinline__ v8f wmma_bf16(v16bf a, v16bf b, v8f c) {
  return __builtin_amdgcn_wmma_f32_16x16x32_bf16(false, a, false, b, (short)0, c,
                                                 false, false);
}

// async 16B global->LDS DMA (gfx1250), tracked by ASYNCcnt
__device__ __forceinline__ void async_b128(const bf16* g, const bf16* l) {
  unsigned lds = (unsigned)(uintptr_t)l;          // low 32 bits = LDS offset
  unsigned long long ga = (unsigned long long)(uintptr_t)g;
  asm volatile("global_load_async_to_lds_b128 %0, %1, off"
               :: "v"(lds), "v"(ga) : "memory");
}
__device__ __forceinline__ void wait_async0() {
  asm volatile("s_wait_asynccnt 0" ::: "memory");
}

// ---------------- RMSNorm -> bf16 ----------------
__global__ __launch_bounds__(256)
void rmsnorm_kernel(const float* __restrict__ x, const float* __restrict__ w,
                    bf16* __restrict__ xn) {
  __shared__ float red[8];
  const int row = blockIdx.x;
  const float* xr = x + (long)row * HDIM;
  float ss = 0.f;
  for (int i = threadIdx.x; i < HDIM; i += 256) { float v = xr[i]; ss += v * v; }
#pragma unroll
  for (int m = 1; m <= 16; m <<= 1) ss += __shfl_xor(ss, m, 32);
  if ((threadIdx.x & 31) == 0) red[threadIdx.x >> 5] = ss;
  __syncthreads();
  float tot = 0.f;
#pragma unroll
  for (int k = 0; k < 8; ++k) tot += red[k];
  const float sc = rsqrtf(tot * (1.0f / HDIM) + 1e-5f);
  for (int i = threadIdx.x; i < HDIM; i += 256)
    xn[(long)row * HDIM + i] = (bf16)(xr[i] * sc * w[i]);
}

// ---------------- f32 -> bf16 cast (vectorized x4) ----------------
__global__ __launch_bounds__(256)
void cvt_bf16_kernel(const float4* __restrict__ in, uint2* __restrict__ out, int n4) {
  int i = blockIdx.x * 256 + threadIdx.x;
  const int stride = gridDim.x * 256;
  for (; i < n4; i += stride) {
    float4 v = in[i];
    union { bf16 b[4]; uint2 u; } o;
    o.b[0] = (bf16)v.x; o.b[1] = (bf16)v.y; o.b[2] = (bf16)v.z; o.b[3] = (bf16)v.w;
    out[i] = o.u;
  }
}

// ---------------- bf16 WMMA GEMM: C[M,N] = A[M,K] * B[N,K]^T ----------------
// async-DMA double-buffered pipeline: DMA(s+1) overlaps WMMA(s)
__global__ __launch_bounds__(256)
void gemm_bf16_kernel(const bf16* __restrict__ A, const bf16* __restrict__ B,
                      float* __restrict__ C, int M, int N, int K) {
  __shared__ __align__(16) bf16 sA[2][128 * 72];
  __shared__ __align__(16) bf16 sB[2][128 * 72];
  const int tid  = threadIdx.x;
  const int lane = tid & 31;
  const int wave = tid >> 5;
  const int wm   = wave >> 2;      // 0..1  (64-row wave tile)
  const int wn   = wave & 3;       // 0..3  (32-col wave tile)
  const int hi   = lane >> 4;
  const int l15  = lane & 15;
  const long m0 = (long)blockIdx.y * 128;
  const long n0 = (long)blockIdx.x * 128;

  v8f acc[4][2];
#pragma unroll
  for (int i = 0; i < 4; ++i)
#pragma unroll
    for (int j = 0; j < 2; ++j) acc[i][j] = (v8f){0,0,0,0,0,0,0,0};

  const int lrow = tid >> 3;         // 0..31
  const int lcol = (tid & 7) * 8;    // bf16 column (16B granule)
  const bf16* gA = A + (m0 + lrow) * (long)K + lcol;
  const bf16* gB = B + (n0 + lrow) * (long)K + lcol;

  auto issue_stage = [&](int buf, int k0) {
#pragma unroll
    for (int r = 0; r < 128; r += 32) {
      async_b128(gA + (long)r * K + k0, &sA[buf][(lrow + r) * 72 + lcol]);
      async_b128(gB + (long)r * K + k0, &sB[buf][(lrow + r) * 72 + lcol]);
    }
  };

  issue_stage(0, 0);
  wait_async0();
  __syncthreads();

  const int nst = K >> 6;
  for (int s = 0; s < nst; ++s) {
    const int cur = s & 1;
    if (s + 1 < nst) issue_stage(1 - cur, (s + 1) << 6);   // overlaps compute
    const bf16* cA = sA[cur];
    const bf16* cB = sB[cur];
#pragma unroll
    for (int ks = 0; ks < 2; ++ks) {
      const int kb = ks * 32 + hi * 8;
      v16bf afr[4], bfr[2];
#pragma unroll
      for (int i = 0; i < 4; ++i)
        afr[i] = ldfrag(&cA[(wm * 64 + i * 16 + l15) * 72 + kb]);
#pragma unroll
      for (int j = 0; j < 2; ++j)
        bfr[j] = ldfrag(&cB[(wn * 32 + j * 16 + l15) * 72 + kb]);
#pragma unroll
      for (int i = 0; i < 4; ++i)
#pragma unroll
        for (int j = 0; j < 2; ++j) acc[i][j] = wmma_bf16(afr[i], bfr[j], acc[i][j]);
    }
    wait_async0();        // next-stage DMA has had the whole compute phase
    __syncthreads();
  }
#pragma unroll
  for (int i = 0; i < 4; ++i)
#pragma unroll
    for (int j = 0; j < 2; ++j)
#pragma unroll
      for (int e = 0; e < 8; ++e) {
        long m = m0 + wm * 64 + i * 16 + e + 8 * hi;
        long n = n0 + wn * 32 + j * 16 + l15;
        C[m * N + n] = acc[i][j][e];
      }
}

// ---------------- RoPE (scale folded into Q) ----------------
__global__ __launch_bounds__(256)
void rope_kernel(const int* __restrict__ positions, const float* __restrict__ qkv,
                 bf16* __restrict__ q_bf, bf16* __restrict__ k_bf,
                 bf16* __restrict__ v_bf) {
  const int t = blockIdx.x;
  const float pos = (float)positions[t];
  const float* row = qkv + (long)t * QKVSZ;
  const float L2TH = 13.287712379549449f;       // log2(10000)
  const float QS = 0.08838834764831845f;        // 1/sqrt(128)
  for (int p = threadIdx.x; p < 2048; p += 256) {   // Q: 32 heads * 64 pairs
    int head = p >> 6, d = p & 63;
    float x1 = row[head * HD + d];
    float x2 = row[head * HD + d + 64];
    float f = pos * exp2f(-(float)(2 * d) * (1.0f / HD) * L2TH);
    float sn, cs; __sincosf(f, &sn, &cs);
    q_bf[(long)t * QSZ + head * HD + d]      = (bf16)((x1 * cs - x2 * sn) * QS);
    q_bf[(long)t * QSZ + head * HD + d + 64] = (bf16)((x2 * cs + x1 * sn) * QS);
  }
  for (int p = threadIdx.x; p < 512; p += 256) {    // K: 8 heads * 64 pairs
    int head = p >> 6, d = p & 63;
    const float* kr = row + QSZ + head * HD;
    float x1 = kr[d], x2 = kr[d + 64];
    float f = pos * exp2f(-(float)(2 * d) * (1.0f / HD) * L2TH);
    float sn, cs; __sincosf(f, &sn, &cs);
    k_bf[((long)t * NKV + head) * HD + d]      = (bf16)(x1 * cs - x2 * sn);
    k_bf[((long)t * NKV + head) * HD + d + 64] = (bf16)(x2 * cs + x1 * sn);
  }
  for (int i = threadIdx.x; i < KVSZ; i += 256)     // V passthrough
    v_bf[(long)t * KVSZ + i] = (bf16)row[QSZ + KVSZ + i];
}

// ---------------- causal flash attention (bf16 WMMA, async K DMA) ----------------
__global__ __launch_bounds__(256)
void attn_kernel(const bf16* __restrict__ q_bf, const bf16* __restrict__ k_bf,
                 const bf16* __restrict__ v_bf, bf16* __restrict__ attn_out) {
  __shared__ __align__(16) bf16 sK[2][32 * 136];    // [kv_tok][hd], padded
  __shared__ __align__(16) bf16 sVt[2][128 * 40];   // [hd][kv_tok], padded
  __shared__ __align__(16) bf16 sP[8][16 * 40];     // per-wave P tile

  const int tid  = threadIdx.x;
  const int lane = tid & 31;
  const int wave = tid >> 5;
  const int hi   = lane >> 4;
  const int l15  = lane & 15;
  const int h    = blockIdx.y;
  const int kvh  = h >> 2;
  const int qb   = blockIdx.x * 128;
  const int q0   = qb + wave * 16;

  // loader geometry
  const int kr = tid >> 4, kc = (tid & 15) * 8;     // K: 2 slots/thread (2nd row +16)
  const int vr0 = tid >> 6, vd = (tid & 63) * 2;    // V: 4 slots/thread (uint)

  auto issue_k = [&](int buf, int kb) {
    async_b128(k_bf + ((long)(kb + kr) * NKV + kvh) * HD + kc,      &sK[buf][kr * 136 + kc]);
    async_b128(k_bf + ((long)(kb + kr + 16) * NKV + kvh) * HD + kc, &sK[buf][(kr + 16) * 136 + kc]);
  };
  // V prefetch: 4 x uint (2 bf16) per thread, rows vr0+{0,8,16,24}, dims vd..vd+1
  auto load_v = [&](int kb, unsigned* reg) {
#pragma unroll
    for (int q = 0; q < 4; ++q)
      reg[q] = *(const unsigned*)(v_bf + ((long)(kb + vr0 + 8 * q) * NKV + kvh) * HD + vd);
  };
  auto store_vt = [&](int buf, const unsigned* reg) {
#pragma unroll
    for (int q = 0; q < 4; ++q) {
      union { unsigned u; bf16 b[2]; } o; o.u = reg[q];
      sVt[buf][vd * 40 + vr0 + 8 * q]       = o.b[0];
      sVt[buf][(vd + 1) * 40 + vr0 + 8 * q] = o.b[1];
    }
  };

  v16bf qf[4];
  {
    const bf16* qrow = q_bf + ((long)(q0 + l15) * NHEAD + h) * HD + hi * 8;
#pragma unroll
    for (int c = 0; c < 4; ++c) qf[c] = ldfrag(qrow + c * 32);
  }
  v8f O[8];
#pragma unroll
  for (int dt = 0; dt < 8; ++dt) O[dt] = (v8f){0,0,0,0,0,0,0,0};
  float mrow[8], lrow[8];
#pragma unroll
  for (int e = 0; e < 8; ++e) { mrow[e] = -1e30f; lrow[e] = 0.f; }

  const int ntiles = (qb + 128) >> 5;
  // prologue: stage tile 0
  {
    unsigned vreg[4];
    issue_k(0, 0);
    load_v(0, vreg);
    store_vt(0, vreg);
    wait_async0();
    __syncthreads();
  }

  for (int jt = 0; jt < ntiles; ++jt) {
    const int kb = jt << 5;
    const int cur = jt & 1;
    unsigned vreg[4];
    const bool has_next = (jt + 1 < ntiles);
    if (has_next) {
      issue_k(1 - cur, kb + 32);       // async DMA overlaps compute
      load_v(kb + 32, vreg);           // global latency hidden under compute
    }

    if (kb <= q0 + 15) {               // wave-uniform causal skip
      v8f s0 = (v8f){0,0,0,0,0,0,0,0}, s1 = (v8f){0,0,0,0,0,0,0,0};
#pragma unroll
      for (int c = 0; c < 4; ++c) {
        const int ko = c * 32 + hi * 8;
        v16bf b0 = ldfrag(&sK[cur][l15 * 136 + ko]);
        v16bf b1 = ldfrag(&sK[cur][(16 + l15) * 136 + ko]);
        s0 = wmma_bf16(qf[c], b0, s0);
        s1 = wmma_bf16(qf[c], b1, s1);
      }
      float p0[8], p1[8];
#pragma unroll
      for (int e = 0; e < 8; ++e) {
        const int qr = q0 + e + 8 * hi;
        float v0 = (kb + l15      <= qr) ? s0[e] : -1e30f;
        float v1 = (kb + 16 + l15 <= qr) ? s1[e] : -1e30f;
        float rm = fmaxf(v0, v1);
#pragma unroll
        for (int msk = 1; msk <= 8; msk <<= 1) rm = fmaxf(rm, __shfl_xor(rm, msk, 32));
        const float mn = fmaxf(mrow[e], rm);
        const float alpha = __expf(mrow[e] - mn);
        const float e0 = __expf(v0 - mn);
        const float e1 = __expf(v1 - mn);
        float rs = e0 + e1;
#pragma unroll
        for (int msk = 1; msk <= 8; msk <<= 1) rs += __shfl_xor(rs, msk, 32);
        lrow[e] = lrow[e] * alpha + rs;
        mrow[e] = mn;
        p0[e] = e0; p1[e] = e1;
#pragma unroll
        for (int dt = 0; dt < 8; ++dt) O[dt][e] *= alpha;
      }
#pragma unroll
      for (int e = 0; e < 8; ++e) {
        const int m = e + 8 * hi;
        sP[wave][m * 40 + l15]      = (bf16)p0[e];
        sP[wave][m * 40 + 16 + l15] = (bf16)p1[e];
      }
      // wave-internal LDS RAW: hardware keeps DS ops in order per wave;
      // stop compiler hoisting + drain DScnt.
      asm volatile("s_wait_dscnt 0" ::: "memory");
      v16bf pf = ldfrag(&sP[wave][l15 * 40 + hi * 8]);
#pragma unroll
      for (int dt = 0; dt < 8; ++dt) {
        v16bf vf = ldfrag(&sVt[cur][(dt * 16 + l15) * 40 + hi * 8]);
        O[dt] = wmma_bf16(pf, vf, O[dt]);
      }
    }

    if (has_next) store_vt(1 - cur, vreg);   // after compute, before end barrier
    wait_async0();
    __syncthreads();
  }
#pragma unroll
  for (int dt = 0; dt < 8; ++dt)
#pragma unroll
    for (int e = 0; e < 8; ++e) {
      long m = q0 + e + 8 * hi;
      int  n = h * HD + dt * 16 + l15;
      attn_out[m * QSZ + n] = (bf16)(O[dt][e] / lrow[e]);
    }
}

// ---------------- launch ----------------
extern "C" void kernel_launch(void* const* d_in, const int* in_sizes, int n_in,
                              void* d_out, int out_size, void* d_ws, size_t ws_size,
                              hipStream_t stream) {
  (void)in_sizes; (void)n_in; (void)out_size; (void)ws_size;
  const int*   positions = (const int*)d_in[0];
  const float* hidden    = (const float*)d_in[1];
  const float* ln_w      = (const float*)d_in[2];
  const float* w_qkv     = (const float*)d_in[3];
  const float* w_o       = (const float*)d_in[4];

  char* ws = (char*)d_ws;
  bf16*  xn_bf   = (bf16*)(ws);                       // 16 MB
  bf16*  wqkv_bf = (bf16*)(ws + 16777216);            // 48 MB
  bf16*  wo_bf   = (bf16*)(ws + 67108864);            // 32 MB
  float* qkv     = (float*)(ws + 100663296);          // 48 MB
  bf16*  q_bf    = (bf16*)(ws + 150994944);           // 16 MB
  bf16*  k_bf    = (bf16*)(ws + 167772160);           // 4 MB
  bf16*  v_bf    = (bf16*)(ws + 171966464);           // 4 MB
  bf16*  attn_bf = (bf16*)(ws + 176160768);           // 16 MB

  rmsnorm_kernel<<<S_LEN, 256, 0, stream>>>(hidden, ln_w, xn_bf);
  cvt_bf16_kernel<<<2048, 256, 0, stream>>>((const float4*)w_qkv, (uint2*)wqkv_bf,
                                            QKVSZ * HDIM / 4);
  cvt_bf16_kernel<<<2048, 256, 0, stream>>>((const float4*)w_o, (uint2*)wo_bf,
                                            HDIM * QSZ / 4);
  gemm_bf16_kernel<<<dim3(QKVSZ / 128, S_LEN / 128), 256, 0, stream>>>(
      xn_bf, wqkv_bf, qkv, S_LEN, QKVSZ, HDIM);
  rope_kernel<<<S_LEN, 256, 0, stream>>>(positions, qkv, q_bf, k_bf, v_bf);
  attn_kernel<<<dim3(S_LEN / 128, NHEAD), 256, 0, stream>>>(q_bf, k_bf, v_bf, attn_bf);
  gemm_bf16_kernel<<<dim3(HDIM / 128, S_LEN / 128), 256, 0, stream>>>(
      attn_bf, wo_bf, (float*)d_out, S_LEN, HDIM, QSZ);
}
